// NET_40913858461753
// MI455X (gfx1250) — matrix operand order repaired and checked by reading, not compiled, not run on hardware
//
#include <hip/hip_runtime.h>
#include <hip/hip_bf16.h>

// ---------------- problem constants ----------------
#define N_NODES 20000
#define N_EDGES 320000
#define IN_FEATS 512
#define HIDDEN 64
#define HEADS 8
#define OUT_DIM 64
#define NEG_SLOPE 0.2f

typedef __attribute__((ext_vector_type(16))) __bf16 v16bf;
typedef __attribute__((ext_vector_type(8)))  __bf16 v8bf;
typedef __attribute__((ext_vector_type(8)))  float  v8f;

// ---------------- helpers ----------------
__device__ __forceinline__ unsigned short f2bf(float f) {
    unsigned u = __float_as_uint(f);
    unsigned r = u + 0x7FFFu + ((u >> 16) & 1u);   // RNE
    return (unsigned short)(r >> 16);
}
__device__ __forceinline__ unsigned flipf(float f) {
    unsigned u = __float_as_uint(f);
    return (u & 0x80000000u) ? ~u : (u | 0x80000000u);
}
__device__ __forceinline__ float unflipf(unsigned v) {
    unsigned u = (v & 0x80000000u) ? (v & 0x7FFFFFFFu) : ~v;
    return __uint_as_float(u);
}

// ---------------- small utility kernels ----------------
__global__ void fill_kernel(unsigned* __restrict__ p, unsigned v, int n) {
    int i = blockIdx.x * blockDim.x + threadIdx.x;
    if (i < n) p[i] = v;
}

__global__ void f32_to_bf16_kernel(const float* __restrict__ in,
                                   unsigned short* __restrict__ out, int n) {
    int i = blockIdx.x * blockDim.x + threadIdx.x;
    if (i < n) out[i] = f2bf(in[i]);
}

// W [K][Nout] f32 (row-major) -> Wt [Nout][K] bf16
__global__ void w_transpose_bf16_kernel(const float* __restrict__ W,
                                        unsigned short* __restrict__ Wt,
                                        int K, int Nout) {
    int i = blockIdx.x * blockDim.x + threadIdx.x;
    if (i < K * Nout) {
        int k = i / Nout;
        int n = i - k * Nout;
        Wt[(size_t)n * K + k] = f2bf(W[i]);
    }
}

// ELU (f32) -> bf16
__global__ void elu_bf16_kernel(const float* __restrict__ in,
                                unsigned short* __restrict__ out, int n) {
    int i = blockIdx.x * blockDim.x + threadIdx.x;
    if (i < n) {
        float x = in[i];
        float y = (x > 0.f) ? x : expm1f(x);
        out[i] = f2bf(y);
    }
}

// ---------------- WMMA GEMM: C[M,Nout] = A[M,K] * Bt[Nout,K]^T ----------------
// A row-major bf16, Bt is B transposed (N-major) bf16, C f32.
// Block = 256 threads = 8 waves; wave w -> rows [by*128 + 16w, +16), cols [bx*64, +64).
__global__ __launch_bounds__(256) void gemm_bf16_wmma_kernel(
    const unsigned short* __restrict__ A,
    const unsigned short* __restrict__ Bt,
    float* __restrict__ C,
    int M, int Nout, int K)
{
    const int lane = threadIdx.x & 31;
    const int wave = threadIdx.x >> 5;
    const int m0 = blockIdx.y * 128 + wave * 16;
    const int n0 = blockIdx.x * 64;

    // A fragment addressing: lane<16 -> row=m0+lane, K 0..15 contiguous;
    //                        lane>=16 -> row=m0+lane-16, K 16..31 contiguous.
    int arow = m0 + (lane & 15);
    if (arow >= M) arow = M - 1;          // clamp; stores are guarded
    const int kbaseA = (lane >> 4) * 16;
    // B fragment: lane<16 -> col n0+lane, K 0..15 contiguous; lane>=16 -> K 16..31.
    const int bcol   = lane & 15;
    const int kbaseB = (lane >> 4) * 16;

    v8f acc0 = {}, acc1 = {}, acc2 = {}, acc3 = {};

    const unsigned short* arow_p = A + (size_t)arow * K;
    const unsigned short* bcol_p = Bt + (size_t)(n0 + bcol) * K;

    for (int k = 0; k < K; k += 32) {
        union { v16bf v; v8bf h[2]; } a;
        const __bf16* ap = (const __bf16*)(arow_p + k + kbaseA);
        a.h[0] = *(const v8bf*)(ap);
        a.h[1] = *(const v8bf*)(ap + 8);

        const __bf16* bp = (const __bf16*)(bcol_p + k + kbaseB);
        v16bf b0 = *(const v16bf*)(bp);
        v16bf b1 = *(const v16bf*)(bp + (size_t)16 * K);
        v16bf b2 = *(const v16bf*)(bp + (size_t)32 * K);
        v16bf b3 = *(const v16bf*)(bp + (size_t)48 * K);

        acc0 = __builtin_amdgcn_wmma_f32_16x16x32_bf16(false, a.v, false, b0, (short)0, acc0, false, false);
        acc1 = __builtin_amdgcn_wmma_f32_16x16x32_bf16(false, a.v, false, b1, (short)0, acc1, false, false);
        acc2 = __builtin_amdgcn_wmma_f32_16x16x32_bf16(false, a.v, false, b2, (short)0, acc2, false, false);
        acc3 = __builtin_amdgcn_wmma_f32_16x16x32_bf16(false, a.v, false, b3, (short)0, acc3, false, false);
    }

    // C/D layout: VGPR r, lanes 0-15 -> (M=r, N=lane); lanes 16-31 -> (M=8+r, N=lane-16).
    const int mtop = m0 + ((lane >> 4) ? 8 : 0);
    const int col0 = n0 + (lane & 15);
    for (int r = 0; r < 8; ++r) {
        int row = mtop + r;
        if (row < M) {
            float* cp = C + (size_t)row * Nout + col0;
            cp[0]  = acc0[r];
            cp[16] = acc1[r];
            cp[32] = acc2[r];
            cp[48] = acc3[r];
        }
    }
}

// ---------------- el/er: per (node, head) dot of feat with al/ar ----------------
// One wave per (n,h); D fixed at 64; lane handles 2 elements, wave-reduce.
__global__ __launch_bounds__(256) void eler_kernel(
    const float* __restrict__ feat,
    const float* __restrict__ al, const float* __restrict__ ar,
    float* __restrict__ el, float* __restrict__ er,
    int Nn, int H)
{
    int gw = blockIdx.x * 8 + (threadIdx.x >> 5);
    int lane = threadIdx.x & 31;
    if (gw >= Nn * H) return;
    int n = gw / H;
    int h = gw - n * H;
    int d = lane * 2;
    const float* f = feat + (size_t)n * (H * 64) + h * 64;
    float f0 = f[d], f1 = f[d + 1];
    float sl = f0 * al[h * 64 + d] + f1 * al[h * 64 + d + 1];
    float sr = f0 * ar[h * 64 + d] + f1 * ar[h * 64 + d + 1];
    for (int off = 16; off > 0; off >>= 1) {
        sl += __shfl_down(sl, off);
        sr += __shfl_down(sr, off);
    }
    if (lane == 0) { el[gw] = sl; er[gw] = sr; }
}

// ---------------- edge softmax phase ----------------
__global__ void edge_max_kernel(const int* __restrict__ src, const int* __restrict__ dst,
                                const float* __restrict__ el, const float* __restrict__ er,
                                unsigned* __restrict__ mbits, int E, int H)
{
    int i = blockIdx.x * blockDim.x + threadIdx.x;
    if (i >= E * H) return;
    int e = i / H, h = i - e * H;
    float s = el[src[e] * H + h] + er[dst[e] * H + h];
    s = (s > 0.f) ? s : NEG_SLOPE * s;
    atomicMax(&mbits[dst[e] * H + h], flipf(s));
}

__global__ void edge_exp_kernel(const int* __restrict__ src, const int* __restrict__ dst,
                                const float* __restrict__ el, const float* __restrict__ er,
                                const unsigned* __restrict__ mbits,
                                float* __restrict__ ex, float* __restrict__ denom,
                                int E, int H)
{
    int i = blockIdx.x * blockDim.x + threadIdx.x;
    if (i >= E * H) return;
    int e = i / H, h = i - e * H;
    int dh = dst[e] * H + h;
    float s = el[src[e] * H + h] + er[dh];
    s = (s > 0.f) ? s : NEG_SLOPE * s;
    float v = __expf(s - unflipf(mbits[dh]));
    ex[i] = v;
    atomicAdd(&denom[dh], v);
}

__global__ void alpha_kernel(const int* __restrict__ dst, const float* __restrict__ denom,
                             float* __restrict__ ex, int E, int H)
{
    int i = blockIdx.x * blockDim.x + threadIdx.x;
    if (i >= E * H) return;
    int e = i / H, h = i - e * H;
    ex[i] = ex[i] / fmaxf(denom[dst[e] * H + h], 1e-9f);
}

// out[dst, :] += alpha[e,h] * feat[src, :]; DTOT = H*64, DSH = log2(DTOT)
__global__ __launch_bounds__(256) void aggregate_kernel(
    const int* __restrict__ src, const int* __restrict__ dst,
    const float* __restrict__ alpha, const float* __restrict__ feat,
    float* __restrict__ out, int E, int H, int DSH)
{
    int i = blockIdx.x * blockDim.x + threadIdx.x;
    int total = E << DSH;
    if (i >= total) return;
    int e = i >> DSH;
    int j = i & ((1 << DSH) - 1);
    int h = j >> 6;
    float a = alpha[e * H + h];
    int dtot = 1 << DSH;
    atomicAdd(&out[(size_t)dst[e] * dtot + j], a * feat[(size_t)src[e] * dtot + j]);
}

// ---------------- launcher ----------------
extern "C" void kernel_launch(void* const* d_in, const int* in_sizes, int n_in,
                              void* d_out, int out_size, void* d_ws, size_t ws_size,
                              hipStream_t stream) {
    const float* features = (const float*)d_in[0];
    const int*   src      = (const int*)d_in[1];
    const int*   dst      = (const int*)d_in[2];
    const float* W0       = (const float*)d_in[3];
    const float* al0      = (const float*)d_in[4];
    const float* ar0      = (const float*)d_in[5];
    const float* W1       = (const float*)d_in[6];
    const float* al1      = (const float*)d_in[7];
    const float* ar1      = (const float*)d_in[8];
    float* out = (float*)d_out;

    const int NF   = N_NODES * IN_FEATS;          // 10,240,000
    const int NH0  = N_NODES * HEADS;             // 160,000
    const int EH0  = N_EDGES * HEADS;             // 2,560,000
    const int NF1  = N_NODES * OUT_DIM;           // 1,280,000

    // workspace carve (everything fits in L2-sized ~143 MB)
    char* ws = (char*)d_ws;
    size_t off = 0;
    auto carve = [&](size_t bytes) -> char* {
        char* p = ws + off;
        off += (bytes + 255) & ~(size_t)255;
        return p;
    };
    unsigned short* Abf  = (unsigned short*)carve((size_t)NF * 2);        // features bf16
    unsigned short* W0t  = (unsigned short*)carve((size_t)512 * 512 * 2); // W0^T bf16
    unsigned short* W1t  = (unsigned short*)carve((size_t)64 * 512 * 2);  // W1^T bf16
    float*  feat0 = (float*)carve((size_t)NF * 4);                        // projection layer 0
    float*  el0   = (float*)carve((size_t)NH0 * 4);
    float*  er0   = (float*)carve((size_t)NH0 * 4);
    unsigned* m0  = (unsigned*)carve((size_t)NH0 * 4);
    float*  den0  = (float*)carve((size_t)NH0 * 4);
    float*  ex0   = (float*)carve((size_t)EH0 * 4);
    float*  hagg  = (float*)carve((size_t)NF * 4);                        // agg output (pre-ELU)
    unsigned short* hbf = (unsigned short*)carve((size_t)NF * 2);         // ELU(h) bf16
    float*  feat1 = (float*)carve((size_t)NF1 * 4);
    float*  el1   = (float*)carve((size_t)N_NODES * 4);
    float*  er1   = (float*)carve((size_t)N_NODES * 4);
    unsigned* m1  = (unsigned*)carve((size_t)N_NODES * 4);
    float*  den1  = (float*)carve((size_t)N_NODES * 4);
    float*  ex1   = (float*)carve((size_t)N_EDGES * 4);
    (void)ws_size; (void)in_sizes; (void)n_in; (void)out_size;

    const int T = 256;
    auto blk = [](int n, int t) { return (n + t - 1) / t; };
    const unsigned NEG_INF_FLIPPED = 0x007FFFFFu;  // flipf(-inf)

    // 0) converts + inits
    f32_to_bf16_kernel<<<blk(NF, T), T, 0, stream>>>(features, Abf, NF);
    w_transpose_bf16_kernel<<<blk(512 * 512, T), T, 0, stream>>>(W0, W0t, 512, 512);
    w_transpose_bf16_kernel<<<blk(512 * 64, T), T, 0, stream>>>(W1, W1t, 512, 64);
    fill_kernel<<<blk(NH0, T), T, 0, stream>>>(m0, NEG_INF_FLIPPED, NH0);
    fill_kernel<<<blk(NH0, T), T, 0, stream>>>((unsigned*)den0, 0u, NH0);
    fill_kernel<<<blk(NF, T), T, 0, stream>>>((unsigned*)hagg, 0u, NF);
    fill_kernel<<<blk(N_NODES, T), T, 0, stream>>>(m1, NEG_INF_FLIPPED, N_NODES);
    fill_kernel<<<blk(N_NODES, T), T, 0, stream>>>((unsigned*)den1, 0u, N_NODES);
    fill_kernel<<<blk(NF1, T), T, 0, stream>>>((unsigned*)out, 0u, NF1);

    // 1) GEMM0: feat0 = features @ W0   (20000x512x512, bf16 WMMA, f32 acc)
    {
        dim3 grid(512 / 64, (N_NODES + 127) / 128);
        gemm_bf16_wmma_kernel<<<grid, 256, 0, stream>>>(Abf, W0t, feat0, N_NODES, 512, 512);
    }
    // 2) el/er layer 0
    eler_kernel<<<blk(NH0, 8), 256, 0, stream>>>(feat0, al0, ar0, el0, er0, N_NODES, HEADS);
    // 3) edge softmax layer 0
    edge_max_kernel<<<blk(EH0, T), T, 0, stream>>>(src, dst, el0, er0, m0, N_EDGES, HEADS);
    edge_exp_kernel<<<blk(EH0, T), T, 0, stream>>>(src, dst, el0, er0, m0, ex0, den0, N_EDGES, HEADS);
    alpha_kernel<<<blk(EH0, T), T, 0, stream>>>(dst, den0, ex0, N_EDGES, HEADS);
    // 4) scatter-add layer 0 (dominant phase; L2-resident)
    aggregate_kernel<<<blk(N_EDGES << 9, T), T, 0, stream>>>(src, dst, ex0, feat0, hagg,
                                                             N_EDGES, HEADS, 9);
    // 5) ELU + bf16 for layer 1 input
    elu_bf16_kernel<<<blk(NF, T), T, 0, stream>>>(hagg, hbf, NF);
    // 6) GEMM1: feat1 = elu(h) @ W1   (20000x512x64)
    {
        dim3 grid(1, (N_NODES + 127) / 128);
        gemm_bf16_wmma_kernel<<<grid, 256, 0, stream>>>(hbf, W1t, feat1, N_NODES, 64, 512);
    }
    // 7) el/er layer 1 (H=1)
    eler_kernel<<<blk(N_NODES, 8), 256, 0, stream>>>(feat1, al1, ar1, el1, er1, N_NODES, 1);
    // 8) edge softmax layer 1
    edge_max_kernel<<<blk(N_EDGES, T), T, 0, stream>>>(src, dst, el1, er1, m1, N_EDGES, 1);
    edge_exp_kernel<<<blk(N_EDGES, T), T, 0, stream>>>(src, dst, el1, er1, m1, ex1, den1, N_EDGES, 1);
    alpha_kernel<<<blk(N_EDGES, T), T, 0, stream>>>(dst, den1, ex1, N_EDGES, 1);
    // 9) scatter-add layer 1 directly into d_out (mean over 1 head == identity)
    aggregate_kernel<<<blk(N_EDGES << 6, T), T, 0, stream>>>(src, dst, ex1, feat1, out,
                                                             N_EDGES, 1, 6);
}
